// TemporalEncoding_nBatch_geometric_attention_78950088835537
// MI455X (gfx1250) — matrix operand order repaired and checked by reading, not compiled, not run on hardware
//
#include <hip/hip_runtime.h>
#include <hip/hip_bf16.h>
#include <math.h>

typedef _Float16 v4h  __attribute__((ext_vector_type(4)));
typedef _Float16 v8h  __attribute__((ext_vector_type(8)));
typedef _Float16 v16h __attribute__((ext_vector_type(16)));
typedef float    v8f  __attribute__((ext_vector_type(8)));

#define D 128
#define ROWS 32    // rows (attention queries) per workgroup = 2 M-tiles
#define PADH 136   // f16 LDS row stride (halves): 272B -> bank-shifted
#define PADF 132   // f32 LDS row stride (floats): 528B -> bank-shifted

// ---- f16 workspace layout (halves) ----
#define WS_WQ   0
#define WS_WK   16384
#define WS_WV   32768
#define WS_W1   49152
#define WS_LIN  65536
#define WS_WOT  81920   // weight_o transposed: 64x128
#define WS_WPT  90112   // weight_p transposed: 64x128
#define WS_TOT  98304

static __device__ __forceinline__ v8f wmma16x16x32(v16h a, v16h b, v8f c) {
  return __builtin_amdgcn_wmma_f32_16x16x32_f16(false, a, false, b, (short)0, c,
                                                false, false);
}

// A fragment (16x32 f16) from LDS row-major tile with PADH stride.
// Lane L: row M=L&15; K run = kbase+16*(L>>4)+0..15 (contiguous, 2 halves/VGPR).
static __device__ __forceinline__ v16h loadA(const _Float16* sA, int lane, int kstep) {
  int m  = lane & 15;
  int k0 = kstep * 32 + 16 * (lane >> 4);
  const v8h lo = *(const v8h*)(sA + m * PADH + k0);
  const v8h hi = *(const v8h*)(sA + m * PADH + k0 + 8);
  v16h a;
#pragma unroll
  for (int i = 0; i < 8; ++i) { a[i] = lo[i]; a[i + 8] = hi[i]; }
  return a;
}

// B fragment (32x16 f16) from row-major f16 weight (ld halves per row).
// Lane L: column N=ntile*16+(L&15); K run = kstep*32 + 16*(L>>4) .. +15 (contiguous).
static __device__ __forceinline__ v16h loadB(const _Float16* W, int lane, int ntile,
                                             int kstep, int ld) {
  int n   = ntile * 16 + (lane & 15);
  int col = kstep * 32 + 16 * (lane >> 4);
  return *(const v16h*)(W + n * ld + col);
}

// C/D layout: reg r, lane L -> M = r + 8*(L>>4), N = ntile*16 + (L&15)
static __device__ __forceinline__ void storeTileF(float* dst, v8f acc, int lane, int ntile) {
  int m0 = 8 * (lane >> 4);
  int n  = ntile * 16 + (lane & 15);
#pragma unroll
  for (int r = 0; r < 8; ++r) dst[(m0 + r) * PADF + n] = acc[r];
}

static __device__ __forceinline__ void storeTileH(_Float16* dst, v8f acc, int lane, int ntile) {
  int m0 = 8 * (lane >> 4);
  int n  = ntile * 16 + (lane & 15);
#pragma unroll
  for (int r = 0; r < 8; ++r) dst[(m0 + r) * PADH + n] = (_Float16)acc[r];
}

// ---------------- weight prep: f32 -> f16 (+ transpose for wo/wp) -------------
__global__ void prep_weights(const float* Wq, const float* Wk, const float* Wv,
                             const float* W1, const float* linW,
                             const float* wo, const float* wp, _Float16* w16) {
  int idx = blockIdx.x * blockDim.x + threadIdx.x;
  if (idx < 81920) {
    int mat = idx >> 14, off = idx & 16383;
    const float* src = (mat == 0) ? Wq : (mat == 1) ? Wk : (mat == 2) ? Wv
                      : (mat == 3) ? W1 : linW;
    w16[idx] = (_Float16)src[off];
  } else if (idx < 90112) {
    int t = idx - 81920, n = t >> 7, k = t & 127;
    w16[idx] = (_Float16)wo[k * 64 + n];          // wo_t[n][k] = wo[k][n]
  } else if (idx < WS_TOT) {
    int t = idx - 90112, n = t >> 7, k = t & 127;
    w16[idx] = (_Float16)wp[k * 64 + n];          // wp_t[n][k] = wp[k][n]
  }
}

// ---------------- main fused kernel: 32 rows per workgroup --------------------
__global__ __launch_bounds__(256)
void geo_attn_main(const float* __restrict__ h_p, const float* __restrict__ h_k,
                   const float* __restrict__ h_o,
                   const float* __restrict__ lin_b, const float* __restrict__ bias,
                   const float* __restrict__ bq, const float* __restrict__ bk,
                   const float* __restrict__ bv, const float* __restrict__ b1,
                   const float* __restrict__ ln_g, const float* __restrict__ ln_b,
                   const int* __restrict__ idx_p, const int* __restrict__ idx_k,
                   const _Float16* __restrict__ w16,
                   float* __restrict__ out_hp, float* __restrict__ out_hk,
                   int Mrows, int Ho_rows) {
  __shared__ int s_idxp[ROWS], s_idxk[ROWS];
  __shared__ _Float16 sA_ho[ROWS * PADH];   // ho_sel (f16), kept all stages
  __shared__ _Float16 sA_hp[ROWS * PADH];   // hp_sel -> ctx -> tanh(t)
  __shared__ _Float16 sA_hk[ROWS * PADH];   // hk_sel -> LN(fusion)
  __shared__ float sQ [ROWS * PADF];        // Q -> fusion -> final out
  __shared__ float sK0[ROWS * PADF];
  __shared__ float sK1[ROWS * PADF];
  __shared__ float sV0[ROWS * PADF];
  __shared__ float sV1[ROWS * PADF];

  const int tid  = threadIdx.x;
  const int lane = tid & 31;
  const int wave = tid >> 5;
  const int tile = blockIdx.x;

  // ---- gather indices ----
  if (tid < ROWS) {
    int g = min(tile * ROWS + tid, Mrows - 1);
    s_idxp[tid] = idx_p[g];
  } else if (tid < 2 * ROWS) {
    int t = tid - ROWS;
    int g = min(tile * ROWS + t, Mrows - 1);
    s_idxk[t] = idx_k[g];
  }
  __syncthreads();

  // ---- gather rows, convert to f16 into LDS (float4 per thread-iter) ----
  for (int e = tid; e < ROWS * 32; e += 256) {
    int m  = e >> 5;
    int c4 = e & 31;                        // float4 index within row
    int ip = s_idxp[m];
    int ik = s_idxk[m];
    int io = min(ip, Ho_rows - 1);          // JAX clamps OOB gathers on h_o
    float4 vo = *(const float4*)(h_o + (size_t)io * D + c4 * 4);
    float4 vp = *(const float4*)(h_p + (size_t)ip * D + c4 * 4);
    float4 vk = *(const float4*)(h_k + (size_t)ik * D + c4 * 4);
    v4h ho4 = { (_Float16)vo.x, (_Float16)vo.y, (_Float16)vo.z, (_Float16)vo.w };
    v4h hp4 = { (_Float16)vp.x, (_Float16)vp.y, (_Float16)vp.z, (_Float16)vp.w };
    v4h hk4 = { (_Float16)vk.x, (_Float16)vk.y, (_Float16)vk.z, (_Float16)vk.w };
    *(v4h*)(sA_ho + m * PADH + c4 * 4) = ho4;
    *(v4h*)(sA_hp + m * PADH + c4 * 4) = hp4;
    *(v4h*)(sA_hk + m * PADH + c4 * 4) = hk4;
  }
  __syncthreads();

  // ---- Q/K/V projections; wave owns N-tile `wave`, 2 M-tiles, B reused ----
  {
    const float bqv = bq[wave * 16 + (lane & 15)];
    v8f accQ0, accQ1;
#pragma unroll
    for (int r = 0; r < 8; ++r) { accQ0[r] = bqv; accQ1[r] = bqv; }
#pragma unroll
    for (int ks = 0; ks < 4; ++ks) {
      v16h b = loadB(w16 + WS_WQ, lane, wave, ks, D);
      accQ0 = wmma16x16x32(loadA(sA_ho,             lane, ks), b, accQ0);
      accQ1 = wmma16x16x32(loadA(sA_ho + 16 * PADH, lane, ks), b, accQ1);
    }
    storeTileF(sQ,             accQ0, lane, wave);
    storeTileF(sQ + 16 * PADF, accQ1, lane, wave);

    const float bkv = bk[wave * 16 + (lane & 15)];
    v8f aK00, aK01, aK10, aK11;             // [key][mtile]
#pragma unroll
    for (int r = 0; r < 8; ++r) { aK00[r] = bkv; aK01[r] = bkv; aK10[r] = bkv; aK11[r] = bkv; }
#pragma unroll
    for (int ks = 0; ks < 4; ++ks) {
      v16h b = loadB(w16 + WS_WK, lane, wave, ks, D);
      aK00 = wmma16x16x32(loadA(sA_hp,             lane, ks), b, aK00);
      aK01 = wmma16x16x32(loadA(sA_hp + 16 * PADH, lane, ks), b, aK01);
      aK10 = wmma16x16x32(loadA(sA_hk,             lane, ks), b, aK10);
      aK11 = wmma16x16x32(loadA(sA_hk + 16 * PADH, lane, ks), b, aK11);
    }
    storeTileF(sK0,             aK00, lane, wave);
    storeTileF(sK0 + 16 * PADF, aK01, lane, wave);
    storeTileF(sK1,             aK10, lane, wave);
    storeTileF(sK1 + 16 * PADF, aK11, lane, wave);

    const float bvv = bv[wave * 16 + (lane & 15)];
    v8f aV00, aV01, aV10, aV11;
#pragma unroll
    for (int r = 0; r < 8; ++r) { aV00[r] = bvv; aV01[r] = bvv; aV10[r] = bvv; aV11[r] = bvv; }
#pragma unroll
    for (int ks = 0; ks < 4; ++ks) {
      v16h b = loadB(w16 + WS_WV, lane, wave, ks, D);
      aV00 = wmma16x16x32(loadA(sA_hp,             lane, ks), b, aV00);
      aV01 = wmma16x16x32(loadA(sA_hp + 16 * PADH, lane, ks), b, aV01);
      aV10 = wmma16x16x32(loadA(sA_hk,             lane, ks), b, aV10);
      aV11 = wmma16x16x32(loadA(sA_hk + 16 * PADH, lane, ks), b, aV11);
    }
    storeTileF(sV0,             aV00, lane, wave);
    storeTileF(sV0 + 16 * PADF, aV01, lane, wave);
    storeTileF(sV1,             aV10, lane, wave);
    storeTileF(sV1 + 16 * PADF, aV11, lane, wave);
  }
  __syncthreads();

  // ---- attention (1 query, 2 keys) + ctx; 8 lanes/row, 16 dims each ----
  {
    int m   = tid >> 3;                     // 0..31
    int sub = tid & 7;
    const float* q  = sQ  + m * PADF + sub * 16;
    const float* k0 = sK0 + m * PADF + sub * 16;
    const float* k1 = sK1 + m * PADF + sub * 16;
    float s0 = 0.f, s1 = 0.f;
#pragma unroll
    for (int i = 0; i < 16; ++i) { s0 += q[i] * k0[i]; s1 += q[i] * k1[i]; }
#pragma unroll
    for (int off = 4; off >= 1; off >>= 1) {
      s0 += __shfl_xor(s0, off, 8);
      s1 += __shfl_xor(s1, off, 8);
    }
    const float scale = 0.08838834764831845f;  // 1/sqrt(128)
    s0 *= scale; s1 *= scale;
    float mx = fmaxf(s0, s1);
    float e0 = __expf(s0 - mx), e1 = __expf(s1 - mx);
    float inv = 1.0f / (e0 + e1);
    float a0 = e0 * inv, a1 = e1 * inv;
    const float* p0 = sV0 + m * PADF + sub * 16;
    const float* p1 = sV1 + m * PADF + sub * 16;
    v8h c0, c1;
#pragma unroll
    for (int i = 0; i < 8; ++i) {
      c0[i] = (_Float16)(a0 * p0[i]     + a1 * p1[i]);
      c1[i] = (_Float16)(a0 * p0[i + 8] + a1 * p1[i + 8]);
    }
    *(v8h*)(sA_hp + m * PADH + sub * 16)     = c0;   // ctx overwrites hp_sel
    *(v8h*)(sA_hp + m * PADH + sub * 16 + 8) = c1;
  }
  __syncthreads();

  // ---- fusion = ctx @ W1^T + b1  (result f32 into sQ) ----
  {
    const float b1v = b1[wave * 16 + (lane & 15)];
    v8f a0, a1;
#pragma unroll
    for (int r = 0; r < 8; ++r) { a0[r] = b1v; a1[r] = b1v; }
#pragma unroll
    for (int ks = 0; ks < 4; ++ks) {
      v16h b = loadB(w16 + WS_W1, lane, wave, ks, D);
      a0 = wmma16x16x32(loadA(sA_hp,             lane, ks), b, a0);
      a1 = wmma16x16x32(loadA(sA_hp + 16 * PADH, lane, ks), b, a1);
    }
    storeTileF(sQ,             a0, lane, wave);
    storeTileF(sQ + 16 * PADF, a1, lane, wave);
  }
  __syncthreads();

  // ---- LayerNorm over D, write normalized f16 into sA_hk ----
  {
    int m   = tid >> 3;
    int sub = tid & 7;
    const float* f = sQ + m * PADF + sub * 16;
    float s = 0.f, s2 = 0.f;
    float v[16];
#pragma unroll
    for (int i = 0; i < 16; ++i) { v[i] = f[i]; s += v[i]; s2 += v[i] * v[i]; }
#pragma unroll
    for (int off = 4; off >= 1; off >>= 1) {
      s  += __shfl_xor(s,  off, 8);
      s2 += __shfl_xor(s2, off, 8);
    }
    float mu  = s  * (1.0f / D);
    float var = s2 * (1.0f / D) - mu * mu;
    float rs  = rsqrtf(var + 1e-5f);
    v8h n0, n1;
#pragma unroll
    for (int i = 0; i < 8; ++i) {
      int d0 = sub * 16 + i, d1 = d0 + 8;
      n0[i] = (_Float16)((v[i]     - mu) * rs * ln_g[d0] + ln_b[d0]);
      n1[i] = (_Float16)((v[i + 8] - mu) * rs * ln_g[d1] + ln_b[d1]);
    }
    *(v8h*)(sA_hk + m * PADH + sub * 16)     = n0;
    *(v8h*)(sA_hk + m * PADH + sub * 16 + 8) = n1;
  }
  __syncthreads();

  // ---- trans_ho (waves 0-3: ho@wo) / trans_pk (waves 4-7: LN@wp); tanh -> sA_hp
  {
    v8f a0, a1;
#pragma unroll
    for (int r = 0; r < 8; ++r) { a0[r] = 0.f; a1[r] = 0.f; }
    const _Float16* A  = (wave < 4) ? sA_ho : sA_hk;
    const _Float16* Wt = (wave < 4) ? (w16 + WS_WOT) : (w16 + WS_WPT);
    int ntile = wave & 3;
#pragma unroll
    for (int ks = 0; ks < 4; ++ks) {
      v16h b = loadB(Wt, lane, ntile, ks, D);
      a0 = wmma16x16x32(loadA(A,             lane, ks), b, a0);
      a1 = wmma16x16x32(loadA(A + 16 * PADH, lane, ks), b, a1);
    }
#pragma unroll
    for (int r = 0; r < 8; ++r) { a0[r] = tanhf(a0[r]); a1[r] = tanhf(a1[r]); }
    storeTileH(sA_hp,             a0, lane, wave);   // concat cols = wave*16..+15
    storeTileH(sA_hp + 16 * PADH, a1, lane, wave);
  }
  __syncthreads();

  // ---- final: leaky_relu(t @ lin_W^T + lin_b) + bias -> sQ ----
  {
    int n = wave * 16 + (lane & 15);
    const float lbv = lin_b[n];
    const float bsv = bias[n];
    v8f a0, a1;
#pragma unroll
    for (int r = 0; r < 8; ++r) { a0[r] = lbv; a1[r] = lbv; }
#pragma unroll
    for (int ks = 0; ks < 4; ++ks) {
      v16h b = loadB(w16 + WS_LIN, lane, wave, ks, D);
      a0 = wmma16x16x32(loadA(sA_hp,             lane, ks), b, a0);
      a1 = wmma16x16x32(loadA(sA_hp + 16 * PADH, lane, ks), b, a1);
    }
#pragma unroll
    for (int r = 0; r < 8; ++r) {
      float x = a0[r]; x = (x > 0.f) ? x : 0.01f * x; a0[r] = x + bsv;
      float y = a1[r]; y = (y > 0.f) ? y : 0.01f * y; a1[r] = y + bsv;
    }
    storeTileF(sQ,             a0, lane, wave);
    storeTileF(sQ + 16 * PADF, a1, lane, wave);
  }
  __syncthreads();

  // ---- scatter: out row -> h_p[idx_p] and h_k[idx_k] copies ----
  {
    int m   = tid >> 3;                     // 0..31
    int sub = tid & 7;                      // 16 floats each
    if (tile * ROWS + m < Mrows) {
      const float4* src = (const float4*)(sQ + m * PADF + sub * 16);
      float4 a = src[0], b = src[1], c = src[2], d = src[3];
      size_t gp = (size_t)s_idxp[m] * D + sub * 16;
      size_t gk = (size_t)s_idxk[m] * D + sub * 16;
      *(float4*)(out_hp + gp)      = a;
      *(float4*)(out_hp + gp + 4)  = b;
      *(float4*)(out_hp + gp + 8)  = c;
      *(float4*)(out_hp + gp + 12) = d;
      *(float4*)(out_hk + gk)      = a;
      *(float4*)(out_hk + gk + 4)  = b;
      *(float4*)(out_hk + gk + 8)  = c;
      *(float4*)(out_hk + gk + 12) = d;
    }
  }
}

extern "C" void kernel_launch(void* const* d_in, const int* in_sizes, int n_in,
                              void* d_out, int out_size, void* d_ws, size_t ws_size,
                              hipStream_t stream) {
  const float* h_p  = (const float*)d_in[0];
  const float* h_k  = (const float*)d_in[1];
  const float* h_o  = (const float*)d_in[2];
  const float* w_o  = (const float*)d_in[3];
  const float* w_p  = (const float*)d_in[4];
  const float* linW = (const float*)d_in[5];
  const float* linb = (const float*)d_in[6];
  const float* bias = (const float*)d_in[7];
  const float* Wq   = (const float*)d_in[8];
  const float* bq   = (const float*)d_in[9];
  const float* Wk   = (const float*)d_in[10];
  const float* bk   = (const float*)d_in[11];
  const float* Wv   = (const float*)d_in[12];
  const float* bv   = (const float*)d_in[13];
  const float* W1   = (const float*)d_in[14];
  const float* b1   = (const float*)d_in[15];
  const float* ln_g = (const float*)d_in[16];
  const float* ln_b = (const float*)d_in[17];
  const int*   idxp = (const int*)d_in[18];
  const int*   idxk = (const int*)d_in[19];

  const int N      = in_sizes[0] / D;     // 300000
  const int Mrows  = in_sizes[18];        // 262144 index pairs
  const int HoRows = in_sizes[2] / D;     // 262144 rows in h_o

  float* out_hp = (float*)d_out;
  float* out_hk = out_hp + (size_t)N * D;

  // outputs start as copies of h_p / h_k (capture-safe async D2D copies)
  hipMemcpyAsync(out_hp, h_p, (size_t)N * D * sizeof(float),
                 hipMemcpyDeviceToDevice, stream);
  hipMemcpyAsync(out_hk, h_k, (size_t)N * D * sizeof(float),
                 hipMemcpyDeviceToDevice, stream);

  _Float16* w16 = (_Float16*)d_ws;        // 196608 B of f16 weights

  prep_weights<<<(WS_TOT + 255) / 256, 256, 0, stream>>>(
      Wq, Wk, Wv, W1, linW, w_o, w_p, w16);

  int tiles = (Mrows + ROWS - 1) / ROWS;
  geo_attn_main<<<tiles, 256, 0, stream>>>(
      h_p, h_k, h_o, linb, bias, bq, bk, bv, b1, ln_g, ln_b,
      idxp, idxk, w16, out_hp, out_hk, Mrows, HoRows);
}
